// MultiHeadAttention_52295521796850
// MI455X (gfx1250) — matrix operand order repaired and checked by reading, not compiled, not run on hardware
//
#include <hip/hip_runtime.h>

// ============================================================================
// MHA forward for MI455X (gfx1250, wave32, WMMA), compile-time-shaped.
//   B=2, S=2048, D=2048, H=16, Dh=128, rot_dim=64.
//   - 4 big GEMMs + attention => compute bound => v_wmma_f32_16x16x32_f16
//   - flash-attention schedule: S x S scores never touch HBM
//   - K/V chunks staged with global_load_async_to_lds_b128 (ASYNCcnt path)
// ============================================================================

constexpr int Bc  = 2;
constexpr int Sc  = 2048;
constexpr int Dc  = 2048;
constexpr int Hc  = 16;
constexpr int Dhc = Dc / Hc;       // 128
constexpr int Mc  = Bc * Sc;       // 4096

typedef __attribute__((ext_vector_type(16))) _Float16 v16h;
typedef __attribute__((ext_vector_type(8)))  float    v8f;

__device__ __forceinline__ v8f wmma_f16(v16h a, v16h b, v8f c) {
  return __builtin_amdgcn_wmma_f32_16x16x32_f16(
      /*neg_a=*/false, a, /*neg_b=*/false, b,
      /*c_mod=*/(short)0, c, /*reuse_a=*/false, /*reuse_b=*/false);
}

// Async global->LDS copy, 16 bytes per lane (GLOBAL_LOAD_ASYNC_TO_LDS_B128,
// GV addressing). vdst = per-lane LDS byte address, vaddr = 64-bit global.
__device__ __forceinline__ void async_copy_b128(void* lds, const void* gptr) {
  unsigned l = (unsigned)(uintptr_t)lds;            // LDS aperture: addr[31:0]
  unsigned long long g = (unsigned long long)(uintptr_t)gptr;
  asm volatile("global_load_async_to_lds_b128 %0, %1, off"
               :: "v"(l), "v"(g) : "memory");
}
__device__ __forceinline__ void wait_async0() {
  asm volatile("s_wait_asynccnt 0" ::: "memory");
}

// --- A fragment (16x32 f16). lane: M = mbase + (lane&15); lanes>=16 add +16
// to K base; VGPR v holds K = 2*v (+1).
__device__ __forceinline__ v16h load_a_frag(const _Float16* p, int ld,
                                            int mbase, int kbase, int lane) {
  v16h a;
  const int m  = mbase + (lane & 15);
  const int ko = kbase + ((lane >> 4) << 4);
  const _Float16* row = p + m * ld;
#pragma unroll
  for (int v = 0; v < 8; ++v) {
    int k = ko + 2 * v;
    a[2 * v]     = row[k];
    a[2 * v + 1] = row[k + 1];
  }
  return a;
}

// --- B fragment (32x16 f16), source row-major along K with row index = n
// (B[k][n] = src[n][k]; used for W and K^T).
__device__ __forceinline__ v16h load_b_rowK(const _Float16* p, int ld,
                                            int nbase, int kbase, int lane) {
  v16h b;
  const int n  = nbase + (lane & 15);
  const int ko = kbase + ((lane >> 4) << 4);
  const _Float16* row = p + n * ld;
#pragma unroll
  for (int v = 0; v < 8; ++v) {
    int k = ko + 2 * v;
    b[2 * v]     = row[k];
    b[2 * v + 1] = row[k + 1];
  }
  return b;
}

// --- B fragment where B[k][n] = src[k][n] (V: row = key, col = dim).
__device__ __forceinline__ v16h load_b_colK(const _Float16* p, int ld,
                                            int nbase, int kbase, int lane) {
  v16h b;
  const int n  = nbase + (lane & 15);
  const int ko = kbase + ((lane >> 4) << 4);
#pragma unroll
  for (int v = 0; v < 8; ++v) {
    int k = ko + 2 * v;
    b[2 * v]     = p[k * ld + n];
    b[2 * v + 1] = p[(k + 1) * ld + n];
  }
  return b;
}

// ============================================================================
// GEMM: C[m,n] = sum_k X[m,k] * W[n,k] + bias[n]   (x @ w.T + b), K = Dc.
// Block: 256 threads = 8 waves; tile 128x128, BK=32. Waves arranged 4x2,
// each wave owns a 32x64 quadrant = 2x4 WMMA accumulators (8 WMMA/k-step).
// MODE 0: f32 out row-major [Mc x Dc] ; MODE 1: f16 out head-major [B,H,S,Dh].
// Staging stays through VGPRs (needs the f32 -> f16 convert).
// ============================================================================
#define GLD 40  // 32 + 8 halves pad

template <int MODE>
__global__ __launch_bounds__(256) void gemm_xwT_kernel(
    const float* __restrict__ X, const float* __restrict__ W,
    const float* __restrict__ bias, float* __restrict__ outf,
    _Float16* __restrict__ outh) {
  __shared__ alignas(16) _Float16 Xs[128 * GLD];
  __shared__ alignas(16) _Float16 Ws[128 * GLD];
  const int tid  = threadIdx.x;
  const int lane = tid & 31;
  const int wid  = tid >> 5;
  const int m0 = blockIdx.x * 128;
  const int n0 = blockIdx.y * 128;
  const int mw = (wid >> 1) << 5;  // 0,32,64,96
  const int nw = (wid & 1) << 6;   // 0,64

  v8f acc[2][4];
  v8f vz = {};
#pragma unroll
  for (int i = 0; i < 2; ++i)
#pragma unroll
    for (int j = 0; j < 4; ++j) acc[i][j] = vz;

  for (int k0 = 0; k0 < Dc; k0 += 32) {
    __syncthreads();
#pragma unroll
    for (int i = tid; i < 128 * 32; i += 256) {  // stage + f32->f16 convert
      int r = i >> 5, c = i & 31;
      Xs[r * GLD + c] = (_Float16)X[(size_t)(m0 + r) * Dc + k0 + c];
      Ws[r * GLD + c] = (_Float16)W[(size_t)(n0 + r) * Dc + k0 + c];
    }
    if (k0 + 32 < Dc) {  // prefetch next K-slab (global_prefetch_b8)
      int r = tid >> 1, c = (tid & 1) << 4;
      __builtin_prefetch(&X[(size_t)(m0 + r) * Dc + k0 + 32 + c], 0, 1);
      __builtin_prefetch(&W[(size_t)(n0 + r) * Dc + k0 + 32 + c], 0, 1);
    }
    __syncthreads();

    v16h af[2], bf[4];
#pragma unroll
    for (int i = 0; i < 2; ++i)
      af[i] = load_a_frag(Xs, GLD, mw + 16 * i, 0, lane);
#pragma unroll
    for (int j = 0; j < 4; ++j)
      bf[j] = load_b_rowK(Ws, GLD, nw + 16 * j, 0, lane);
#pragma unroll
    for (int i = 0; i < 2; ++i)
#pragma unroll
      for (int j = 0; j < 4; ++j) acc[i][j] = wmma_f16(af[i], bf[j], acc[i][j]);
  }

  // Epilogue. C/D layout: M = r + 8*(lane>=16), N = lane&15.
  const int rowoff = (lane >> 4) << 3;
  const int nlane  = lane & 15;
#pragma unroll
  for (int i = 0; i < 2; ++i)
#pragma unroll
    for (int j = 0; j < 4; ++j)
#pragma unroll
      for (int r = 0; r < 8; ++r) {
        int row = m0 + mw + 16 * i + rowoff + r;
        int col = n0 + nw + 16 * j + nlane;
        float val = acc[i][j][r] + bias[col];
        if (MODE == 0) {
          outf[(size_t)row * Dc + col] = val;
        } else {
          int bb = row / Sc, ss = row & (Sc - 1);
          int hh = col / Dhc, dd = col & (Dhc - 1);
          outh[(((size_t)(bb * Hc + hh)) * Sc + ss) * Dhc + dd] = (_Float16)val;
        }
      }
}

// ============================================================================
// RoPE on f16 Q,K (head-major [BH,S,Dh]); rot_dim = Dh/2 = 64 -> 32 pairs.
// theta(p,s) = s * 10000^(-p/32); interleaved pair rotation.
// ============================================================================
constexpr int ROT_HALF = (Dhc / 2) / 2;  // 32

__global__ __launch_bounds__(256) void rope_kernel(_Float16* __restrict__ Q,
                                                   _Float16* __restrict__ K) {
  int idx = blockIdx.x * blockDim.x + threadIdx.x;  // Bc*Hc*Sc*ROT_HALF threads
  int p  = idx & (ROT_HALF - 1);
  int s  = (idx / ROT_HALF) & (Sc - 1);
  int bh = idx / (ROT_HALF * Sc);
  size_t base = ((size_t)bh * Sc + s) * Dhc + 2 * p;
  float theta = (float)s * __powf(10000.0f, -(float)p / (float)ROT_HALF);
  float sn, cs;
  __sincosf(theta, &sn, &cs);
  float q0 = (float)Q[base], q1 = (float)Q[base + 1];
  Q[base]     = (_Float16)(q0 * cs - q1 * sn);
  Q[base + 1] = (_Float16)(q1 * cs + q0 * sn);
  float k0 = (float)K[base], k1 = (float)K[base + 1];
  K[base]     = (_Float16)(k0 * cs - k1 * sn);
  K[base + 1] = (_Float16)(k1 * cs + k0 * sn);
}

// ============================================================================
// Flash attention: block = 128 threads (4 waves), one (b,h), 64 q-rows;
// wave w owns q-rows [16w,16w+16). Keys streamed in 64-wide chunks, staged
// with async global->LDS B128 copies (ASYNCcnt). Online softmax in f32
// registers; P re-fragmented via per-wave LDS for the P@V WMMA.
// ============================================================================
#define FLD 136  // 128 + 8 halves pad (row pitch 272 B, 16B-aligned)
#define PLD 72   // 64 + 8

__global__ __launch_bounds__(128) void flash_attn_kernel(
    const _Float16* __restrict__ Qh, const _Float16* __restrict__ Kh,
    const _Float16* __restrict__ Vh, float* __restrict__ O,
    const int* __restrict__ prefix_ptr) {
  __shared__ alignas(16) _Float16 Qs[64 * FLD];
  __shared__ alignas(16) _Float16 Ks[64 * FLD];
  __shared__ alignas(16) _Float16 Vs[64 * FLD];
  __shared__ alignas(16) _Float16 Ps[4 * 16 * PLD];

  const int tid  = threadIdx.x;
  const int lane = tid & 31;
  const int wid  = tid >> 5;
  const int qbase = blockIdx.x * 64;
  const int bh = blockIdx.y;
  const int b = bh / Hc, h = bh & (Hc - 1);
  const int prefix = *prefix_ptr;
  const size_t hoff = (size_t)bh * Sc * Dhc;
  const _Float16* Qg = Qh + hoff;
  const _Float16* Kg = Kh + hoff;
  const _Float16* Vg = Vh + hoff;

  // Stage Q tile once: 64x128 f16 = 1024 x 16B chunks, async to LDS.
  for (int c = tid; c < 1024; c += 128) {
    int r = c >> 4, c8 = (c & 15) << 3;  // row, half-offset (8 halves = 16B)
    async_copy_b128(&Qs[r * FLD + c8], Qg + (size_t)(qbase + r) * Dhc + c8);
  }
  wait_async0();
  __syncthreads();

  v16h qa[4];
#pragma unroll
  for (int ks = 0; ks < 4; ++ks)
    qa[ks] = load_a_frag(Qs, FLD, wid * 16, ks * 32, lane);

  float mstate[8], lstate[8];
  v8f oacc[8];
  v8f vz = {};
#pragma unroll
  for (int r = 0; r < 8; ++r) { mstate[r] = -1e30f; lstate[r] = 0.0f; }
#pragma unroll
  for (int j = 0; j < 8; ++j) oacc[j] = vz;

  const float scale = rsqrtf((float)Dhc);
  const int rowoff = (lane >> 4) << 3;
  const int nlane  = lane & 15;
  _Float16* Pw = Ps + wid * 16 * PLD;

  int kend = qbase + 64 > prefix ? qbase + 64 : prefix;  // causal | prefix
  if (kend > Sc) kend = Sc;

  for (int kk0 = 0; kk0 < kend; kk0 += 64) {
    __syncthreads();  // previous chunk's consumers done before overwrite
    for (int c = tid; c < 1024; c += 128) {  // async-stage K,V chunk
      int r = c >> 4, c8 = (c & 15) << 3;
      async_copy_b128(&Ks[r * FLD + c8], Kg + (size_t)(kk0 + r) * Dhc + c8);
      async_copy_b128(&Vs[r * FLD + c8], Vg + (size_t)(kk0 + r) * Dhc + c8);
    }
    if (kk0 + 64 < kend) {  // prefetch next chunk toward L2
      int r = tid >> 1, c = (tid & 1) << 6;
      __builtin_prefetch(&Kg[(size_t)(kk0 + 64 + r) * Dhc + c], 0, 1);
      __builtin_prefetch(&Vg[(size_t)(kk0 + 64 + r) * Dhc + c], 0, 1);
    }
    wait_async0();
    __syncthreads();

    // S(16x64) = Q(16x128) * K^T(128x64): 4 k-steps x 4 n-subtiles.
    // B fragments batched per k-step so DS loads overlap.
    v8f sacc[4];
#pragma unroll
    for (int j = 0; j < 4; ++j) sacc[j] = vz;
#pragma unroll
    for (int ks = 0; ks < 4; ++ks) {
      v16h kb[4];
#pragma unroll
      for (int j = 0; j < 4; ++j)
        kb[j] = load_b_rowK(Ks, FLD, j * 16, ks * 32, lane);
#pragma unroll
      for (int j = 0; j < 4; ++j) sacc[j] = wmma_f16(qa[ks], kb[j], sacc[j]);
    }

    // Mask + scale.
#pragma unroll
    for (int j = 0; j < 4; ++j) {
      int key = kk0 + j * 16 + nlane;
#pragma unroll
      for (int r = 0; r < 8; ++r) {
        int qrow = qbase + wid * 16 + rowoff + r;
        bool ok = (key <= qrow) || (key < prefix);
        sacc[j][r] = ok ? sacc[j][r] * scale : -1e30f;
      }
    }

    // Online softmax per row (row spread over 16 lanes).
#pragma unroll
    for (int r = 0; r < 8; ++r) {
      float mx = fmaxf(fmaxf(sacc[0][r], sacc[1][r]),
                       fmaxf(sacc[2][r], sacc[3][r]));
      for (int off = 8; off >= 1; off >>= 1)
        mx = fmaxf(mx, __shfl_xor(mx, off, 32));
      float mnew = fmaxf(mstate[r], mx);
      float corr = __expf(mstate[r] - mnew);
      float psum = 0.0f;
#pragma unroll
      for (int j = 0; j < 4; ++j) {
        float pv = __expf(sacc[j][r] - mnew);
        sacc[j][r] = pv;
        psum += pv;
      }
      for (int off = 8; off >= 1; off >>= 1) psum += __shfl_xor(psum, off, 32);
      lstate[r] = lstate[r] * corr + psum;
      mstate[r] = mnew;
#pragma unroll
      for (int j2 = 0; j2 < 8; ++j2) oacc[j2][r] *= corr;
    }

    // Re-fragment P through per-wave LDS (C-layout -> A-layout).
#pragma unroll
    for (int j = 0; j < 4; ++j)
#pragma unroll
      for (int r = 0; r < 8; ++r)
        Pw[(rowoff + r) * PLD + j * 16 + nlane] = (_Float16)sacc[j][r];
    __syncthreads();  // uniform across block; makes P visible wave-wide

    // O(16x128) += P(16x64) * V(64x128): 2 k-steps x 8 n-subtiles,
    // V fragments batched in groups of 4.
#pragma unroll
    for (int ks2 = 0; ks2 < 2; ++ks2) {
      v16h pa = load_a_frag(Pw, PLD, 0, ks2 * 32, lane);
#pragma unroll
      for (int g = 0; g < 2; ++g) {
        v16h vb[4];
#pragma unroll
        for (int j = 0; j < 4; ++j)
          vb[j] = load_b_colK(Vs, FLD, (g * 4 + j) * 16, ks2 * 32, lane);
#pragma unroll
        for (int j = 0; j < 4; ++j)
          oacc[g * 4 + j] = wmma_f16(pa, vb[j], oacc[g * 4 + j]);
      }
    }
  }

  // Normalize and write O (f32, [B,S,D] layout for the output GEMM).
#pragma unroll
  for (int j2 = 0; j2 < 8; ++j2)
#pragma unroll
    for (int r = 0; r < 8; ++r) {
      int qrow = qbase + wid * 16 + rowoff + r;
      int d = j2 * 16 + nlane;
      O[((size_t)b * Sc + qrow) * Dc + h * Dhc + d] = oacc[j2][r] / lstate[r];
    }
}

// ============================================================================
extern "C" void kernel_launch(void* const* d_in, const int* in_sizes, int n_in,
                              void* d_out, int out_size, void* d_ws,
                              size_t ws_size, hipStream_t stream) {
  (void)in_sizes; (void)n_in; (void)out_size; (void)ws_size;

  const float* q   = (const float*)d_in[0];
  const float* WQw = (const float*)d_in[1];
  const float* WQb = (const float*)d_in[2];
  const float* WKw = (const float*)d_in[3];
  const float* WKb = (const float*)d_in[4];
  const float* WVw = (const float*)d_in[5];
  const float* WVb = (const float*)d_in[6];
  const float* WOw = (const float*)d_in[7];
  const float* WOb = (const float*)d_in[8];
  const int* prefix = (const int*)d_in[9];
  float* out = (float*)d_out;

  // Workspace: Qh/Kh/Vh f16 head-major (16.8 MB each) + O f32 (33.6 MB).
  _Float16* Qh = (_Float16*)d_ws;
  _Float16* Kh = Qh + (size_t)Mc * Dc;
  _Float16* Vh = Kh + (size_t)Mc * Dc;
  float* Obuf  = (float*)(Vh + (size_t)Mc * Dc);

  dim3 gGemm(Mc / 128, Dc / 128);
  dim3 blkG(256);

  // QKV projections (f32 -> f16 head-major [B,H,S,Dh]).
  gemm_xwT_kernel<1><<<gGemm, blkG, 0, stream>>>(q, WQw, WQb, nullptr, Qh);
  gemm_xwT_kernel<1><<<gGemm, blkG, 0, stream>>>(q, WKw, WKb, nullptr, Kh);
  gemm_xwT_kernel<1><<<gGemm, blkG, 0, stream>>>(q, WVw, WVb, nullptr, Vh);

  // RoPE on Q and K.
  const int ropeTotal = Bc * Hc * Sc * ROT_HALF;  // 2,097,152
  rope_kernel<<<ropeTotal / 256, 256, 0, stream>>>(Qh, Kh);

  // Flash attention -> Obuf f32 [B,S,D].
  dim3 gAttn(Sc / 64, Bc * Hc);
  flash_attn_kernel<<<gAttn, dim3(128), 0, stream>>>(Qh, Kh, Vh, Obuf, prefix);

  // Output projection -> d_out f32.
  gemm_xwT_kernel<0><<<gGemm, blkG, 0, stream>>>(Obuf, WOw, WOb, out, nullptr);
}
